// DenseGeneral_79207786873200
// MI455X (gfx1250) — 8-bit matrix operand order repaired and checked by reading, compiled, not run on hardware
//
#include <hip/hip_runtime.h>
#include <hip/hip_bf16.h>
#include <math.h>

typedef __attribute__((ext_vector_type(16))) int   v16i;
typedef __attribute__((ext_vector_type(8)))  float v8f;

#define E4M3_MAX 448.0f

#if __has_builtin(__builtin_amdgcn_global_load_async_to_lds_b128) && \
    __has_builtin(__builtin_amdgcn_s_wait_asynccnt)
#define HAVE_ASYNC_LDS 1
#else
#define HAVE_ASYNC_LDS 0
#endif

__device__ __forceinline__ float fast_tanhf(float x) {
#if __has_builtin(__builtin_amdgcn_tanhf)
    return __builtin_amdgcn_tanhf(x);
#elif __has_builtin(__builtin_amdgcn_tanh_f32)
    return __builtin_amdgcn_tanh_f32(x);
#else
    return tanhf(x);
#endif
}

__device__ __forceinline__ unsigned f32_to_e4m3_byte(float x) {
    unsigned b   = __float_as_uint(x);
    unsigned sgn = (b >> 31) << 7;
    unsigned ab  = b & 0x7fffffffu;
    if (ab > 0x7f800000u) return sgn | 0x7fu;
    if (ab == 0u)         return sgn;
    int e8 = (int)(ab >> 23) - 127 + 7;
    if (e8 >= 1) {
        unsigned full = (ab & 0x7fffffu) | 0x800000u;
        unsigned q    = full >> 20;
        unsigned rem  = full & 0xfffffu;
        if (rem > 0x80000u || (rem == 0x80000u && (q & 1u))) q++;
        if (q == 16u) { q = 8u; e8++; }
        if (e8 > 15) return sgn | 0x7eu;
        return sgn | ((unsigned)e8 << 3) | (q & 7u);
    } else {
        float t = __uint_as_float(ab) * 512.0f;
        int   n = (int)rintf(t);
        if (n >= 8) return sgn | 0x08u;
        return sgn | (unsigned)n;
    }
}

__device__ __forceinline__ float gelu_tanh(float v) {
    float c = 0.7978845608028654f * (v + 0.044715f * v * v * v);
    return 0.5f * v * (1.0f + fast_tanhf(c));
}

__global__ void init_ws_kernel(unsigned* amax_bits) {
    amax_bits[0] = 0u;
    amax_bits[1] = 0u;
}

__global__ void amax_kernel(const float4* __restrict__ x, int n4,
                            unsigned* __restrict__ amax_bits) {
    float m = 0.0f;
    int stride = gridDim.x * blockDim.x;
    for (int i = blockIdx.x * blockDim.x + threadIdx.x; i < n4; i += stride) {
        float4 v = x[i];
        m = fmaxf(m, fmaxf(fmaxf(fabsf(v.x), fabsf(v.y)),
                           fmaxf(fabsf(v.z), fabsf(v.w))));
    }
    __shared__ float red[256];
    red[threadIdx.x] = m;
    __syncthreads();
    for (int s = 128; s > 0; s >>= 1) {
        if (threadIdx.x < s)
            red[threadIdx.x] = fmaxf(red[threadIdx.x], red[threadIdx.x + s]);
        __syncthreads();
    }
    if (threadIdx.x == 0) atomicMax(amax_bits, __float_as_uint(red[0]));
}

__global__ void scale_kernel(const unsigned* __restrict__ amax_bits,
                             const float* __restrict__ in_scale,
                             const float* __restrict__ k_scale,
                             float* __restrict__ scales) {
    float ax = __uint_as_float(amax_bits[0]);
    float ak = __uint_as_float(amax_bits[1]);
    scales[0] = (ax > 0.0f && ax <= 3.402823466e38f) ? ax * (1.0f / E4M3_MAX) : in_scale[0];
    scales[1] = (ak > 0.0f && ak <= 3.402823466e38f) ? ak * (1.0f / E4M3_MAX) : k_scale[0];
}

__global__ void quant_x_kernel(const float4* __restrict__ x,
                               unsigned* __restrict__ q, int n4,
                               const float* __restrict__ scales) {
    const float inv = 1.0f / scales[0];
    int stride = gridDim.x * blockDim.x;
    for (int i = blockIdx.x * blockDim.x + threadIdx.x; i < n4; i += stride) {
        float4 v = x[i];
        unsigned b0 = f32_to_e4m3_byte(fminf(fmaxf(v.x * inv, -E4M3_MAX), E4M3_MAX));
        unsigned b1 = f32_to_e4m3_byte(fminf(fmaxf(v.y * inv, -E4M3_MAX), E4M3_MAX));
        unsigned b2 = f32_to_e4m3_byte(fminf(fmaxf(v.z * inv, -E4M3_MAX), E4M3_MAX));
        unsigned b3 = f32_to_e4m3_byte(fminf(fmaxf(v.w * inv, -E4M3_MAX), E4M3_MAX));
        q[i] = b0 | (b1 << 8) | (b2 << 16) | (b3 << 24);
    }
}

#define WT  64
#define WTS 80

__global__ __launch_bounds__(256) void quant_w_kernel(
        const float* __restrict__ w, unsigned char* __restrict__ qkT,
        int K, int N, const float* __restrict__ scales) {
    __shared__ __align__(16) unsigned char t[WT * WTS];
    const float inv = 1.0f / scales[1];
    const int k0 = blockIdx.x * WT;
    const int n0 = blockIdx.y * WT;
    const int tid = threadIdx.x;

    #pragma unroll
    for (int p = 0; p < 4; ++p) {
        int j   = tid + (p << 8);
        int row = j >> 4;
        int c4  = j & 15;
        float4 v = *(const float4*)(w + (size_t)(k0 + row) * N + n0 + c4 * 4);
        t[(c4 * 4 + 0) * WTS + row] =
            (unsigned char)f32_to_e4m3_byte(fminf(fmaxf(v.x * inv, -E4M3_MAX), E4M3_MAX));
        t[(c4 * 4 + 1) * WTS + row] =
            (unsigned char)f32_to_e4m3_byte(fminf(fmaxf(v.y * inv, -E4M3_MAX), E4M3_MAX));
        t[(c4 * 4 + 2) * WTS + row] =
            (unsigned char)f32_to_e4m3_byte(fminf(fmaxf(v.z * inv, -E4M3_MAX), E4M3_MAX));
        t[(c4 * 4 + 3) * WTS + row] =
            (unsigned char)f32_to_e4m3_byte(fminf(fmaxf(v.w * inv, -E4M3_MAX), E4M3_MAX));
    }
    __syncthreads();
    {
        int nloc = tid >> 2;
        int ch   = (tid & 3) << 4;
        uint4 d = *(const uint4*)(t + nloc * WTS + ch);
        *(uint4*)(qkT + (size_t)(n0 + nloc) * K + k0 + ch) = d;
    }
}

#define MT  128
#define NT  128
#define KT  128
#define LDT 144

#if HAVE_ASYNC_LDS
typedef int v4i_g __attribute__((vector_size(16)));
typedef __attribute__((address_space(1))) v4i_g* gvp_t;
typedef __attribute__((address_space(3))) v4i_g* lvp_t;
#endif

__global__ __launch_bounds__(256) void gemm_fp8_gelu_kernel(
        const unsigned char* __restrict__ qx,
        const unsigned char* __restrict__ qkT,
        const float* __restrict__ scales,
        const float* __restrict__ bias,
        float* __restrict__ out,
        int M, int N, int K) {
    __shared__ __align__(16) unsigned char smA[2][MT * LDT];
    __shared__ __align__(16) unsigned char smB[2][NT * LDT];

    const int tid  = threadIdx.x;
    const int lane = tid & 31;
    const int wave = tid >> 5;
    const int wm = wave >> 1;
    const int wn = wave & 1;
    const int lr = lane & 15;
    const int lg = lane >> 4;

    const int M0 = blockIdx.y * MT;
    const int N0 = blockIdx.x * NT;
    const unsigned char* gA = qx  + (size_t)M0 * K;
    const unsigned char* gB = qkT + (size_t)N0 * K;

    const v8f vzero = {0.f, 0.f, 0.f, 0.f, 0.f, 0.f, 0.f, 0.f};
    v8f acc[2][4];
    #pragma unroll
    for (int i = 0; i < 2; ++i)
        #pragma unroll
        for (int j = 0; j < 4; ++j) acc[i][j] = vzero;

    const int kSteps = K / KT;

#if HAVE_ASYNC_LDS
    {
        #pragma unroll
        for (int p = 0; p < 4; ++p) {
            int u = tid + (p << 8);
            int row = u >> 3, col = (u & 7) << 4;
            __builtin_amdgcn_global_load_async_to_lds_b128(
                (gvp_t)(gA + (size_t)row * K + col),
                (lvp_t)(&smA[0][row * LDT + col]), 0, 0);
            __builtin_amdgcn_global_load_async_to_lds_b128(
                (gvp_t)(gB + (size_t)row * K + col),
                (lvp_t)(&smB[0][row * LDT + col]), 0, 0);
        }
    }
#else
    uint4 va[4], vb[4];
    #pragma unroll
    for (int p = 0; p < 4; ++p) {
        int u = tid + (p << 8);
        int row = u >> 3, col = (u & 7) << 4;
        va[p] = *(const uint4*)(gA + (size_t)row * K + col);
        vb[p] = *(const uint4*)(gB + (size_t)row * K + col);
    }
#endif

    #pragma unroll 2
    for (int ks = 0; ks < kSteps; ++ks) {
        const int cur = ks & 1;

#if HAVE_ASYNC_LDS
        __builtin_amdgcn_s_wait_asynccnt(0);
        __syncthreads();
        if (ks + 1 < kSteps) {
            const int k1 = (ks + 1) * KT;
            #pragma unroll
            for (int p = 0; p < 4; ++p) {
                int u = tid + (p << 8);
                int row = u >> 3, col = (u & 7) << 4;
                __builtin_amdgcn_global_load_async_to_lds_b128(
                    (gvp_t)(gA + (size_t)row * K + k1 + col),
                    (lvp_t)(&smA[cur ^ 1][row * LDT + col]), 0, 0);
                __builtin_amdgcn_global_load_async_to_lds_b128(
                    (gvp_t)(gB + (size_t)row * K + k1 + col),
                    (lvp_t)(&smB[cur ^ 1][row * LDT + col]), 0, 0);
            }
        }
#else
        __syncthreads();
        #pragma unroll
        for (int p = 0; p < 4; ++p) {
            int u = tid + (p << 8);
            int row = u >> 3, col = (u & 7) << 4;
            *(uint4*)(&smA[cur][row * LDT + col]) = va[p];
            *(uint4*)(&smB[cur][row * LDT + col]) = vb[p];
        }
        __syncthreads();
        if (ks + 1 < kSteps) {
            const int k1 = (ks + 1) * KT;
            #pragma unroll
            for (int p = 0; p < 4; ++p) {
                int u = tid + (p << 8);
                int row = u >> 3, col = (u & 7) << 4;
                va[p] = *(const uint4*)(gA + (size_t)row * K + k1 + col);
                vb[p] = *(const uint4*)(gB + (size_t)row * K + k1 + col);
            }
        }
#endif

        v16i af[2];
        #pragma unroll
        for (int fm = 0; fm < 2; ++fm) {
            const unsigned char* base =
                &smA[cur][(wm * 32 + fm * 16 + lr) * LDT + lg * 16];
            #pragma unroll
            for (int p = 0; p < 8; ++p) {
                unsigned long long d = *(const unsigned long long*)(base + (p >> 1) * 32 + (p & 1) * 8);
                af[fm][2 * p]     = (int)(unsigned)d;
                af[fm][2 * p + 1] = (int)(unsigned)(d >> 32);
            }
        }
        v16i bf[4];
        #pragma unroll
        for (int fn = 0; fn < 4; ++fn) {
            const unsigned char* base =
                &smB[cur][(wn * 64 + fn * 16 + lr) * LDT + lg * 16];
            #pragma unroll
            for (int q = 0; q < 4; ++q) {
                uint4 d = *(const uint4*)(base + q * 32);
                bf[fn][4 * q + 0] = (int)d.x;
                bf[fn][4 * q + 1] = (int)d.y;
                bf[fn][4 * q + 2] = (int)d.z;
                bf[fn][4 * q + 3] = (int)d.w;
            }
        }

        #pragma unroll
        for (int fm = 0; fm < 2; ++fm)
            #pragma unroll
            for (int fn = 0; fn < 4; ++fn)
                acc[fm][fn] = __builtin_amdgcn_wmma_f32_16x16x128_fp8_fp8(
                    af[fm], bf[fn], (short)0, acc[fm][fn], false, false);
    }

    const float sxk = scales[0] * scales[1];
    #pragma unroll
    for (int fn = 0; fn < 4; ++fn) {
        const int n = N0 + wn * 64 + fn * 16 + lr;
        const float bb = __bfloat162float(__float2bfloat16(bias[n]));
        #pragma unroll
        for (int fm = 0; fm < 2; ++fm) {
            const int mbase = M0 + wm * 32 + fm * 16 + lg * 8;
            #pragma unroll
            for (int e = 0; e < 8; ++e) {
                float v = acc[fm][fn][e] * sxk + bb;
                out[(size_t)(mbase + e) * N + n] = gelu_tanh(v);
            }
        }
    }
}

extern "C" void kernel_launch(void* const* d_in, const int* in_sizes, int n_in,
                              void* d_out, int out_size, void* d_ws, size_t ws_size,
                              hipStream_t stream) {
    const float* inputs   = (const float*)d_in[0];
    const float* kernelw  = (const float*)d_in[1];
    const float* bias     = (const float*)d_in[2];
    const float* in_scale = (const float*)d_in[3];
    const float* k_scale  = (const float*)d_in[5];

    const int F = in_sizes[2];
    const int D = in_sizes[1] / F;
    const int M = in_sizes[0] / D;
    const int N = F;
    const int K = D;

    unsigned char* ws        = (unsigned char*)d_ws;
    unsigned*      amax_bits = (unsigned*)ws;
    float*         scales    = (float*)(ws + 64);
    unsigned char* qx        = ws + 256;
    unsigned char* qkT       = ws + 256 + (size_t)M * K;

    init_ws_kernel<<<1, 1, 0, stream>>>(amax_bits);
    amax_kernel<<<1024, 256, 0, stream>>>((const float4*)inputs,  (M * K) / 4, amax_bits + 0);
    amax_kernel<<<1024, 256, 0, stream>>>((const float4*)kernelw, (K * N) / 4, amax_bits + 1);
    scale_kernel<<<1, 1, 0, stream>>>(amax_bits, in_scale, k_scale, scales);

    quant_x_kernel<<<2048, 256, 0, stream>>>((const float4*)inputs,
                                             (unsigned*)qx, (M * K) / 4, scales);
    dim3 gw(K / WT, N / WT);
    quant_w_kernel<<<gw, 256, 0, stream>>>(kernelw, qkT, K, N, scales);

    dim3 gg(N / NT, M / MT);
    gemm_fp8_gelu_kernel<<<gg, 256, 0, stream>>>(qx, qkT, scales, bias,
                                                 (float*)d_out, M, N, K);
}
